// RSSM_50422916055094
// MI455X (gfx1250) — matrix operand order repaired and checked by reading, not compiled, not run on hardware
//
#include <hip/hip_runtime.h>
#include <hip/hip_bf16.h>

// ---------------- problem dims ----------------
#define B_   256
#define S_   512
#define HID_ 512

typedef __attribute__((ext_vector_type(16))) __bf16 v16bf;
typedef __attribute__((ext_vector_type(8)))  __bf16 v8bf;
typedef __attribute__((ext_vector_type(8)))  float  v8f;

// All pointers the persistent kernel needs (<< 4KB kernarg limit).
struct KParams {
  const float *emb, *act, *noise;
  const __bf16 *preW[5];  const float *preB[5],  *preG[4],  *preBe[4];
  const __bf16 *Wih, *Whh; const float *bih, *bhh;
  const __bf16 *postW[5]; const float *postB[5], *postG[4], *postBe[4];
  const __bf16 *priW[5];  const float *priB[5],  *priG[4],  *priBe[4];
  float *out_h, *out_s, *out_zmu, *out_zsg, *out_hmu, *out_hsg;
};

// ---- one 16x16 output tile accumulated over K (K multiple of 32) ----
// A: LDS activations, row-major [16][as] bf16 (as*2 bytes 16B aligned)
// W: global weights,  row-major [dout_pad][wstride] bf16, 32B aligned rows
template<int K>
__device__ __forceinline__ v8f wmma_acc(v8f acc, const __bf16* A, int as,
                                        const __bf16* W, int wstride, int n0,
                                        int lane) {
  const int nl = lane & 15;        // A row m  /  B col n
  const int hs = lane >> 4;        // A K-half /  B K-group
  const __bf16* arow = A + nl * as;
  const __bf16* wrow = W + (size_t)(n0 + nl) * wstride;
#pragma unroll
  for (int k0 = 0; k0 < K; k0 += 32) {
    v8bf lo = *(const v8bf*)(arow + k0 + 8 * hs);
    v8bf hi = *(const v8bf*)(arow + k0 + 16 + 8 * hs);
    v16bf a;
#pragma unroll
    for (int i = 0; i < 8; i++) { a[i] = lo[i]; a[i + 8] = hi[i]; }
    v8bf blo = *(const v8bf*)(wrow + k0 + 8 * hs);
    v8bf bhi = *(const v8bf*)(wrow + k0 + 16 + 8 * hs);
    v16bf b;
#pragma unroll
    for (int i = 0; i < 8; i++) { b[i] = blo[i]; b[i + 8] = bhi[i]; }
    acc = __builtin_amdgcn_wmma_f32_16x16x32_bf16(false, a, false, b,
                                                  (short)0, acc, false, false);
  }
  return acc;
}

// GEMM layer: y[16][256] = A @ W^T + bias  (waves stride over N tiles)
template<int K>
__device__ __forceinline__ void gemm_layer(const __bf16* A, int as,
                                           const __bf16* W, int wstride,
                                           const float* bias, int dout,
                                           int Ntiles, float* y, int tid) {
  const int w = tid >> 5, lane = tid & 31, nl = lane & 15, hs = lane >> 4;
  for (int tile = w; tile < Ntiles; tile += 8) {
    const int n0 = tile * 16;
    v8f acc = {};
    acc = wmma_acc<K>(acc, A, as, W, wstride, n0, lane);
    const float bz = (n0 + nl) < dout ? bias[n0 + nl] : 0.f;
#pragma unroll
    for (int j = 0; j < 8; j++) y[(j + 8 * hs) * 256 + n0 + nl] = acc[j] + bz;
  }
}

// LayerNorm(+ReLU) over feature dim: y[16][256] f32 -> x[16][256] bf16
// (zero-pads x columns [dout, dpad) for the next GEMM's padded K)
__device__ __forceinline__ void ln_relu(const float* y, __bf16* x, float* red,
                                        float2* stat, const float* g,
                                        const float* be, int dout, int dpad,
                                        int tid) {
  const int row = tid >> 4, c = tid & 15;
  float s = 0.f, ss = 0.f;
  for (int cc = c; cc < dout; cc += 16) {
    float v = y[row * 256 + cc]; s += v; ss += v * v;
  }
  red[row * 16 + c] = s; red[256 + row * 16 + c] = ss;
  __syncthreads();
  if (c == 0) {
    float s1 = 0.f, s2 = 0.f;
    for (int i = 0; i < 16; i++) { s1 += red[row * 16 + i]; s2 += red[256 + row * 16 + i]; }
    const float mean = s1 / (float)dout;
    const float var  = s2 / (float)dout - mean * mean;
    stat[row] = make_float2(mean, rsqrtf(var + 1e-5f));
  }
  __syncthreads();
  const float mean = stat[row].x, rstd = stat[row].y;
  for (int cc = c; cc < dpad; cc += 16) {
    float o = 0.f;
    if (cc < dout) {
      float v = y[row * 256 + cc];
      o = fmaxf((v - mean) * rstd * g[cc] + be[cc], 0.f);
    }
    x[row * 256 + cc] = (__bf16)o;
  }
}

// ---------------- persistent RSSM kernel: 16 blocks x 16 batch rows ----------------
extern "C" __global__ void __launch_bounds__(256, 1) rssm_kernel(KParams P) {
  __shared__ __bf16 s_h[16 * 512];    // hidden state h_{t-1} (bf16, GEMM A)
  __shared__ __bf16 s_emb[16 * 512];  // input embedding for step t
  __shared__ __bf16 s_x[16 * 256];    // layer input (bf16, GEMM A)
  __shared__ float  s_y[16 * 256];    // layer output (f32 staging)
  __shared__ float  s_red[512];
  __shared__ float2 s_stat[16];
  __shared__ __bf16 s_smp[16 * 64];   // carried stochastic sample

  const int tid = threadIdx.x, w = tid >> 5, lane = tid & 31;
  const int nl = lane & 15, hs = lane >> 4;
  const int r0 = blockIdx.x * 16;

  for (int i = tid; i < 16 * 512; i += 256) s_h[i] = (__bf16)0.f;
  for (int i = tid; i < 16 * 64; i += 256)  s_smp[i] = (__bf16)0.f;
  v8f h_reg[4] = {};   // this wave's 4 hidden column tiles, f32, live all 512 steps
  __syncthreads();

  for (int t = 0; t < S_; t++) {
    // ---- phase 0: load embedding, build pre-input [smp(64) | a_prev(2) | pad->96]
    for (int i = tid; i < 16 * 512; i += 256) {
      int r = i >> 9, c = i & 511;
      s_emb[i] = (__bf16)P.emb[(((size_t)(r0 + r)) * S_ + t) * 512 + c];
    }
    for (int i = tid; i < 16 * 96; i += 256) {
      int r = i / 96, c = i - 96 * r; float v = 0.f;
      if (c < 64)                 v = (float)s_smp[r * 64 + c];
      else if (c < 66 && t > 0)   v = P.act[(((size_t)(r0 + r)) * S_ + (t - 1)) * 2 + (c - 64)];
      s_x[r * 256 + c] = (__bf16)v;
    }
    __syncthreads();

    // ---- pre rep: ANL(96p->256), ANL(256->128), ANL(128->128), ANL(128->66), Lin(96p->66)
    gemm_layer<96>(s_x, 256, P.preW[0], 96, P.preB[0], 256, 16, s_y, tid);  __syncthreads();
    ln_relu(s_y, s_x, s_red, s_stat, P.preG[0], P.preBe[0], 256, 256, tid); __syncthreads();
    gemm_layer<256>(s_x, 256, P.preW[1], 256, P.preB[1], 128, 8, s_y, tid); __syncthreads();
    ln_relu(s_y, s_x, s_red, s_stat, P.preG[1], P.preBe[1], 128, 128, tid); __syncthreads();
    gemm_layer<128>(s_x, 256, P.preW[2], 128, P.preB[2], 128, 8, s_y, tid); __syncthreads();
    ln_relu(s_y, s_x, s_red, s_stat, P.preG[2], P.preBe[2], 128, 128, tid); __syncthreads();
    gemm_layer<128>(s_x, 256, P.preW[3], 128, P.preB[3], 66, 5, s_y, tid);  __syncthreads();
    ln_relu(s_y, s_x, s_red, s_stat, P.preG[3], P.preBe[3], 66, 96, tid);   __syncthreads();
    gemm_layer<96>(s_x, 256, P.preW[4], 96, P.preB[4], 66, 5, s_y, tid);    __syncthreads();
    for (int i = tid; i < 16 * 96; i += 256) {      // GRU input xg (bf16, pad->96)
      int r = i / 96, c = i - 96 * r;
      s_x[r * 256 + c] = (__bf16)((c < 66) ? s_y[r * 256 + c] : 0.f);
    }
    __syncthreads();

    // ---- GRU cell: gi = xg@Wih^T+bih, gh = h@Whh^T+bhh ; gates r,z,n
    // wave w owns hidden tiles {w, w+8, w+16, w+24}; old h read from s_h, new h kept in regs
#pragma unroll
    for (int it = 0; it < 4; it++) {
      const int c0 = (w + 8 * it) * 16;
      v8f ir = {}, iz = {}, inn = {}, hr = {}, hz = {}, hn = {};
      ir  = wmma_acc<96>(ir,  s_x, 256, P.Wih, 96, c0, lane);
      iz  = wmma_acc<96>(iz,  s_x, 256, P.Wih, 96, 512 + c0, lane);
      inn = wmma_acc<96>(inn, s_x, 256, P.Wih, 96, 1024 + c0, lane);
      hr  = wmma_acc<512>(hr, s_h, 512, P.Whh, 512, c0, lane);
      hz  = wmma_acc<512>(hz, s_h, 512, P.Whh, 512, 512 + c0, lane);
      hn  = wmma_acc<512>(hn, s_h, 512, P.Whh, 512, 1024 + c0, lane);
      const float bir = P.bih[c0 + nl], biz = P.bih[512 + c0 + nl], bin = P.bih[1024 + c0 + nl];
      const float bhr = P.bhh[c0 + nl], bhz = P.bhh[512 + c0 + nl], bhn = P.bhh[1024 + c0 + nl];
#pragma unroll
      for (int j = 0; j < 8; j++) {
        float r = 1.f / (1.f + __expf(-(ir[j] + bir + hr[j] + bhr)));
        float z = 1.f / (1.f + __expf(-(iz[j] + biz + hz[j] + bhz)));
        float nn = tanhf(inn[j] + bin + r * (hn[j] + bhn));
        h_reg[it][j] = (1.f - z) * nn + z * h_reg[it][j];
      }
    }
    __syncthreads();   // everyone done reading old s_h
#pragma unroll
    for (int it = 0; it < 4; it++) {
      const int c0 = (w + 8 * it) * 16;
#pragma unroll
      for (int j = 0; j < 8; j++) {
        const int m = j + 8 * hs; const float hv = h_reg[it][j];
        s_h[m * 512 + c0 + nl] = (__bf16)hv;
        P.out_h[(((size_t)(r0 + m)) * S_ + t) * 512 + c0 + nl] = hv;
      }
    }
    __syncthreads();

    // ---- prior rep on h (K=512): ANL(512->256), ANL(256->128), ANL x2, Lin(128->128)
    gemm_layer<512>(s_h, 512, P.priW[0], 512, P.priB[0], 256, 16, s_y, tid); __syncthreads();
    ln_relu(s_y, s_x, s_red, s_stat, P.priG[0], P.priBe[0], 256, 256, tid);  __syncthreads();
    gemm_layer<256>(s_x, 256, P.priW[1], 256, P.priB[1], 128, 8, s_y, tid);  __syncthreads();
    ln_relu(s_y, s_x, s_red, s_stat, P.priG[1], P.priBe[1], 128, 128, tid);  __syncthreads();
    gemm_layer<128>(s_x, 256, P.priW[2], 128, P.priB[2], 128, 8, s_y, tid);  __syncthreads();
    ln_relu(s_y, s_x, s_red, s_stat, P.priG[2], P.priBe[2], 128, 128, tid);  __syncthreads();
    gemm_layer<128>(s_x, 256, P.priW[3], 128, P.priB[3], 128, 8, s_y, tid);  __syncthreads();
    ln_relu(s_y, s_x, s_red, s_stat, P.priG[3], P.priBe[3], 128, 128, tid);  __syncthreads();
    gemm_layer<128>(s_x, 256, P.priW[4], 128, P.priB[4], 128, 8, s_y, tid);  __syncthreads();
    {
      const int r = tid >> 4, c = tid & 15;
      for (int st = c; st < 64; st += 16) {
        const size_t o = (((size_t)(r0 + r)) * S_ + t) * 64 + st;
        P.out_hmu[o] = s_y[r * 256 + st];
        P.out_hsg[o] = s_y[r * 256 + 64 + st];
      }
    }
    __syncthreads();

    // ---- post rep on [h | emb] (K=1024 split 512+512)
    for (int tile = w; tile < 16; tile += 8) {
      const int n0 = tile * 16;
      v8f acc = {};
      acc = wmma_acc<512>(acc, s_h,   512, P.postW[0],       1024, n0, lane);
      acc = wmma_acc<512>(acc, s_emb, 512, P.postW[0] + 512, 1024, n0, lane);
      const float bz = P.postB[0][n0 + nl];
#pragma unroll
      for (int j = 0; j < 8; j++) s_y[(j + 8 * hs) * 256 + n0 + nl] = acc[j] + bz;
    }
    __syncthreads();
    ln_relu(s_y, s_x, s_red, s_stat, P.postG[0], P.postBe[0], 256, 256, tid); __syncthreads();
    gemm_layer<256>(s_x, 256, P.postW[1], 256, P.postB[1], 128, 8, s_y, tid); __syncthreads();
    ln_relu(s_y, s_x, s_red, s_stat, P.postG[1], P.postBe[1], 128, 128, tid); __syncthreads();
    gemm_layer<128>(s_x, 256, P.postW[2], 128, P.postB[2], 128, 8, s_y, tid); __syncthreads();
    ln_relu(s_y, s_x, s_red, s_stat, P.postG[2], P.postBe[2], 128, 128, tid); __syncthreads();
    gemm_layer<128>(s_x, 256, P.postW[3], 128, P.postB[3], 128, 8, s_y, tid); __syncthreads();
    ln_relu(s_y, s_x, s_red, s_stat, P.postG[3], P.postBe[3], 128, 128, tid); __syncthreads();
    gemm_layer<128>(s_x, 256, P.postW[4], 128, P.postB[4], 128, 8, s_y, tid); __syncthreads();
    {
      const int r = tid >> 4, c = tid & 15;
      for (int st = c; st < 64; st += 16) {
        const size_t o = (((size_t)(r0 + r)) * S_ + t) * 64 + st;
        const float mu = s_y[r * 256 + st];
        const float ls = s_y[r * 256 + 64 + st];
        const float nz = P.noise[(((size_t)t) * B_ + (r0 + r)) * 64 + st];
        const float sm = mu + __expf(ls) * nz;
        P.out_zmu[o] = mu; P.out_zsg[o] = ls; P.out_s[o] = sm;
        s_smp[r * 64 + st] = (__bf16)sm;
      }
    }
    __syncthreads();
  }
}

// ---------------- weight f32 -> padded bf16 conversion ----------------
struct ConvDesc { const float* src; int dout, din, dpad, kpad; long long off; };
struct ConvTab  { ConvDesc d[17]; };

extern "C" __global__ void conv_w(ConvTab T, __bf16* dst) {
  const ConvDesc cd = T.d[blockIdx.y];
  const long long tot = (long long)cd.dpad * cd.kpad;
  for (long long i = (long long)blockIdx.x * blockDim.x + threadIdx.x; i < tot;
       i += (long long)gridDim.x * blockDim.x) {
    const int nrow = (int)(i / cd.kpad), k = (int)(i % cd.kpad);
    float v = (nrow < cd.dout && k < cd.din) ? cd.src[(long long)nrow * cd.din + k] : 0.f;
    dst[cd.off + i] = (__bf16)v;
  }
}

// ---------------- host launch ----------------
extern "C" void kernel_launch(void* const* d_in, const int* in_sizes, int n_in,
                              void* d_out, int out_size, void* d_ws, size_t ws_size,
                              hipStream_t stream) {
  (void)in_sizes; (void)n_in; (void)out_size; (void)ws_size;
  // {d_in idx, dout, din, dout_pad, K_pad} — flatten order of setup_inputs()
  static const int WT[17][5] = {
    {2, 256, 66, 256, 96},  {6, 128, 256, 128, 256}, {10, 128, 128, 128, 128},
    {14, 66, 128, 80, 128}, {18, 66, 66, 80, 96},
    {20, 1536, 66, 1536, 96}, {21, 1536, 512, 1536, 512},
    {24, 256, 1024, 256, 1024}, {28, 128, 256, 128, 256}, {32, 128, 128, 128, 128},
    {36, 128, 128, 128, 128},  {40, 128, 128, 128, 128},
    {42, 256, 512, 256, 512},  {46, 128, 256, 128, 256}, {50, 128, 128, 128, 128},
    {54, 128, 128, 128, 128},  {58, 128, 128, 128, 128}};
  ConvTab T; long long off = 0; long long offs[17];
  for (int i = 0; i < 17; i++) {
    T.d[i].src  = (const float*)d_in[WT[i][0]];
    T.d[i].dout = WT[i][1]; T.d[i].din = WT[i][2];
    T.d[i].dpad = WT[i][3]; T.d[i].kpad = WT[i][4];
    T.d[i].off = off; offs[i] = off;
    off += (long long)WT[i][3] * WT[i][4];   // ~1.58M elems -> ~3.2MB bf16
  }
  __bf16* wsb = (__bf16*)d_ws;
  conv_w<<<dim3(256, 17), 256, 0, stream>>>(T, wsb);

  KParams P{};
  P.emb   = (const float*)d_in[0];
  P.act   = (const float*)d_in[1];
  P.noise = (const float*)d_in[60];
  for (int i = 0; i < 5; i++) {
    P.preW[i]  = wsb + offs[i];
    P.postW[i] = wsb + offs[7 + i];
    P.priW[i]  = wsb + offs[12 + i];
  }
  P.Wih = wsb + offs[5]; P.Whh = wsb + offs[6];

  static const int preB[5] = {3, 7, 11, 15, 19}, preG[4] = {4, 8, 12, 16}, preE[4] = {5, 9, 13, 17};
  static const int posB[5] = {25, 29, 33, 37, 41}, posG[4] = {26, 30, 34, 38}, posE[4] = {27, 31, 35, 39};
  static const int priB[5] = {43, 47, 51, 55, 59}, priG[4] = {44, 48, 52, 56}, priE[4] = {45, 49, 53, 57};
  for (int i = 0; i < 5; i++) {
    P.preB[i]  = (const float*)d_in[preB[i]];
    P.postB[i] = (const float*)d_in[posB[i]];
    P.priB[i]  = (const float*)d_in[priB[i]];
  }
  for (int i = 0; i < 4; i++) {
    P.preG[i]  = (const float*)d_in[preG[i]];  P.preBe[i]  = (const float*)d_in[preE[i]];
    P.postG[i] = (const float*)d_in[posG[i]];  P.postBe[i] = (const float*)d_in[posE[i]];
    P.priG[i]  = (const float*)d_in[priG[i]];  P.priBe[i]  = (const float*)d_in[priE[i]];
  }
  P.bih = (const float*)d_in[22]; P.bhh = (const float*)d_in[23];

  float* out = (float*)d_out;
  const size_t BS = (size_t)B_ * S_;
  P.out_h   = out;
  P.out_s   = out + BS * 512;
  P.out_zmu = P.out_s   + BS * 64;
  P.out_zsg = P.out_zmu + BS * 64;
  P.out_hmu = P.out_zsg + BS * 64;
  P.out_hsg = P.out_hmu + BS * 64;

  rssm_kernel<<<16, 256, 0, stream>>>(P);
}
